// AbstractKWTA_64845416235063
// MI455X (gfx1250) — matrix operand order repaired and checked by reading, not compiled, not run on hardware
//
#include <hip/hip_runtime.h>
#include <math.h>

// Problem constants: x[B=64, C=2048, T=512], w[N=1024, C=2048] -> out[B, N=1024, T=512]
#define BB 64
#define CC 2048
#define TT 512
#define NN 1024

// LDS row stride for bf16 tile rows: 40 bf16 = 80B -> 16B aligned b128 gathers,
// and 20*row mod 64 banks distinct across 16 rows (near conflict-free).
#define LDSTH 40

typedef __attribute__((ext_vector_type(16))) __bf16 v16bf;
typedef __attribute__((ext_vector_type(8)))  float  v8f;

// Native vector type matching the async-LDS builtin's parameter type
// (diagnostic showed: __attribute__((__vector_size__(4*sizeof(int)))) int *).
typedef int v4i __attribute__((vector_size(16)));
typedef __attribute__((address_space(1))) v4i gv4i;  // global-space 16B vector
typedef __attribute__((address_space(3))) v4i lv4i;  // LDS-space 16B vector

#if defined(__has_builtin)
#if __has_builtin(__builtin_amdgcn_global_load_async_to_lds_b128)
#define HAVE_ASYNC_LDS 1
#endif
#if __has_builtin(__builtin_amdgcn_s_wait_asynccnt)
#define WAIT_ASYNC(n) __builtin_amdgcn_s_wait_asynccnt(n)
#else
#define WAIT_ASYNC(n) asm volatile("s_wait_asynccnt %0" ::"i"(n) : "memory")
#endif
#endif

// round-to-nearest-even fp32 -> bf16 bit pattern (low 16 bits)
__device__ __forceinline__ unsigned bf16_rn(float f) {
  unsigned x = __float_as_uint(f);
  return (x + 0x7FFFu + ((x >> 16) & 1u)) >> 16;
}

// ---------------------------------------------------------------------------
// Prep kernel A: split weight into hi/lo bf16 planes (same [n][c] layout).
// ---------------------------------------------------------------------------
__global__ __launch_bounds__(256) void split_w(const float* __restrict__ w,
                                               unsigned short* __restrict__ whi,
                                               unsigned short* __restrict__ wlo) {
  size_t i4 = ((size_t)blockIdx.x * 256 + threadIdx.x) * 4;  // grid covers NN*CC/4
  float4 v = *(const float4*)(w + i4);
  ushort4 H, L;
  {
    unsigned h;
    h = bf16_rn(v.x); H.x = (unsigned short)h; L.x = (unsigned short)bf16_rn(v.x - __uint_as_float(h << 16));
    h = bf16_rn(v.y); H.y = (unsigned short)h; L.y = (unsigned short)bf16_rn(v.y - __uint_as_float(h << 16));
    h = bf16_rn(v.z); H.z = (unsigned short)h; L.z = (unsigned short)bf16_rn(v.z - __uint_as_float(h << 16));
    h = bf16_rn(v.w); H.w = (unsigned short)h; L.w = (unsigned short)bf16_rn(v.w - __uint_as_float(h << 16));
  }
  *(ushort4*)(whi + i4) = H;
  *(ushort4*)(wlo + i4) = L;
}

// ---------------------------------------------------------------------------
// Prep kernel B: transpose + split x[b, c, t] fp32 -> xT_hi/lo[b_local, t, c] bf16.
// ---------------------------------------------------------------------------
__global__ __launch_bounds__(256) void xsplit(const float* __restrict__ x,
                                              unsigned short* __restrict__ xhi,
                                              unsigned short* __restrict__ xlo,
                                              int bbase) {
  __shared__ float tile[64 * 65];
  const int tid = threadIdx.x;
  const int t0 = blockIdx.x * 64;   // T/64 = 8
  const int c0 = blockIdx.y * 64;   // C/64 = 32
  const int bl = blockIdx.z;
  const int bg = bbase + bl;

#pragma unroll
  for (int r = 0; r < 4; ++r) {
    int idx = r * 256 + tid;
    int c = idx >> 4;
    int t4 = (idx & 15) << 2;
    float4 v = *(const float4*)(x + (size_t)(bg * CC + c0 + c) * TT + t0 + t4);
    tile[(t4 + 0) * 65 + c] = v.x;
    tile[(t4 + 1) * 65 + c] = v.y;
    tile[(t4 + 2) * 65 + c] = v.z;
    tile[(t4 + 3) * 65 + c] = v.w;
  }
  __syncthreads();
#pragma unroll
  for (int r = 0; r < 4; ++r) {
    int idx = r * 256 + tid;
    int t = idx >> 4;
    int c4 = (idx & 15) << 2;
    ushort4 H, L;
    float f; unsigned h;
    f = tile[t * 65 + c4 + 0]; h = bf16_rn(f); H.x = (unsigned short)h; L.x = (unsigned short)bf16_rn(f - __uint_as_float(h << 16));
    f = tile[t * 65 + c4 + 1]; h = bf16_rn(f); H.y = (unsigned short)h; L.y = (unsigned short)bf16_rn(f - __uint_as_float(h << 16));
    f = tile[t * 65 + c4 + 2]; h = bf16_rn(f); H.z = (unsigned short)h; L.z = (unsigned short)bf16_rn(f - __uint_as_float(h << 16));
    f = tile[t * 65 + c4 + 3]; h = bf16_rn(f); H.w = (unsigned short)h; L.w = (unsigned short)bf16_rn(f - __uint_as_float(h << 16));
    size_t dst = ((size_t)bl * TT + t0 + t) * CC + c0 + c4;
    *(ushort4*)(xhi + dst) = H;
    *(ushort4*)(xlo + dst) = L;
  }
}

// ---------------------------------------------------------------------------
// Kernel 1: z[b,t,n] = sum_i w[n,i]*x[b,i,t], bf16x3 compensated WMMA.
// 128(t) x 128(n) block tile, K-step 32, DOUBLE-BUFFERED LDS, async
// global->LDS staging when the toolchain exposes it.
// ---------------------------------------------------------------------------
__global__ __launch_bounds__(256) void kwta_gemm_bf16x3(
    const unsigned short* __restrict__ xhi, const unsigned short* __restrict__ xlo,
    const unsigned short* __restrict__ whi, const unsigned short* __restrict__ wlo,
    float* __restrict__ z, int bbase) {
  __shared__ __align__(16) unsigned short AsH[2][128 * LDSTH];
  __shared__ __align__(16) unsigned short AsL[2][128 * LDSTH];
  __shared__ __align__(16) unsigned short BsH[2][128 * LDSTH];
  __shared__ __align__(16) unsigned short BsL[2][128 * LDSTH];

  const int tid   = threadIdx.x;
  const int lane  = tid & 31;
  const int wv    = tid >> 5;
  const int waveM = wv >> 2;    // 0..1
  const int waveN = wv & 3;     // 0..3
  const int lrow  = lane & 15;
  const int lhalf = lane >> 4;  // 0/1 half-wave
  const int n0 = blockIdx.x * 128;
  const int t0 = blockIdx.y * 128;
  const int bl = blockIdx.z;    // local batch (xT buffers)
  const int bg = bbase + bl;    // global batch (z)

  const int srow = tid >> 2;        // 0..63 staging row
  const int sseg = (tid & 3) * 8;   // bf16 k-offset 0/8/16/24 (16B segments)

  const v8f vzero = {0.f, 0.f, 0.f, 0.f, 0.f, 0.f, 0.f, 0.f};
  v8f acc[4][2];
#pragma unroll
  for (int mt = 0; mt < 4; ++mt)
#pragma unroll
    for (int nt = 0; nt < 2; ++nt) acc[mt][nt] = vzero;

  union Frag { uint4 q[2]; v16bf v; };

  // ---- compute phase on one LDS buffer: b128 gathers -> 24 WMMAs ----
  auto compute = [&](int bufi) {
    v16bf ahi[4], alo[4], bhi[2], blo[2];
    const int kb = lhalf * 8;  // A: low lanes K 0-7/16-23, high lanes 8-15/24-31
#pragma unroll
    for (int mt = 0; mt < 4; ++mt) {
      int base = (waveM * 64 + mt * 16 + lrow) * LDSTH + kb;
      Frag H, L;
      H.q[0] = *(const uint4*)&AsH[bufi][base];
      H.q[1] = *(const uint4*)&AsH[bufi][base + 16];
      L.q[0] = *(const uint4*)&AsL[bufi][base];
      L.q[1] = *(const uint4*)&AsL[bufi][base + 16];
      ahi[mt] = H.v;
      alo[mt] = L.v;
    }
    const int kbB = lhalf * 8;  // B: low lanes K 0-7/16-23, high lanes 8-15/24-31
#pragma unroll
    for (int nt = 0; nt < 2; ++nt) {
      int base = (waveN * 32 + nt * 16 + lrow) * LDSTH + kbB;
      Frag H, L;
      H.q[0] = *(const uint4*)&BsH[bufi][base];
      H.q[1] = *(const uint4*)&BsH[bufi][base + 16];
      L.q[0] = *(const uint4*)&BsL[bufi][base];
      L.q[1] = *(const uint4*)&BsL[bufi][base + 16];
      bhi[nt] = H.v;
      blo[nt] = L.v;
    }
#pragma unroll
    for (int mt = 0; mt < 4; ++mt)
#pragma unroll
      for (int nt = 0; nt < 2; ++nt) {
        acc[mt][nt] = __builtin_amdgcn_wmma_f32_16x16x32_bf16(
            false, ahi[mt], false, bhi[nt], (short)0, acc[mt][nt], false, false);
        acc[mt][nt] = __builtin_amdgcn_wmma_f32_16x16x32_bf16(
            false, ahi[mt], false, blo[nt], (short)0, acc[mt][nt], false, false);
        acc[mt][nt] = __builtin_amdgcn_wmma_f32_16x16x32_bf16(
            false, alo[mt], false, bhi[nt], (short)0, acc[mt][nt], false, false);
      }
  };

  const int NK = CC / 32;  // 64 K-steps

#ifdef HAVE_ASYNC_LDS
  // ---- async path: GLOBAL_LOAD_ASYNC_TO_LDS_B128, ASYNCcnt-tracked ----
  auto stage_async = [&](int bufi, int kk) {
#pragma unroll
    for (int h = 0; h < 2; ++h) {
      size_t offA = ((size_t)bl * TT + t0 + srow + h * 64) * CC + kk + sseg;
      size_t offB = (size_t)(n0 + srow + h * 64) * CC + kk + sseg;
      int li = (srow + h * 64) * LDSTH + sseg;
      __builtin_amdgcn_global_load_async_to_lds_b128(
          (gv4i*)(xhi + offA), (lv4i*)&AsH[bufi][li], 0, 0);
      __builtin_amdgcn_global_load_async_to_lds_b128(
          (gv4i*)(xlo + offA), (lv4i*)&AsL[bufi][li], 0, 0);
      __builtin_amdgcn_global_load_async_to_lds_b128(
          (gv4i*)(whi + offB), (lv4i*)&BsH[bufi][li], 0, 0);
      __builtin_amdgcn_global_load_async_to_lds_b128(
          (gv4i*)(wlo + offB), (lv4i*)&BsL[bufi][li], 0, 0);
    }
  };

  stage_async(0, 0);
  for (int i = 0; i < NK; ++i) {
    int cur = i & 1;
    if (i + 1 < NK) {
      stage_async(cur ^ 1, (i + 1) * 32);
      WAIT_ASYNC(8);  // in-order: current tile's 8 copies done, next 8 in flight
    } else {
      WAIT_ASYNC(0);
    }
    __syncthreads();  // all waves' current tile resident
    compute(cur);
    __syncthreads();  // readers done before next iteration's async overwrites
  }
#else
  // ---- sync path: prefetch to regs, store after compute, 1 barrier/step ----
  uint4 gah[2], gal[2], gbh[2], gbl[2];
  auto gload = [&](int kk) {
#pragma unroll
    for (int h = 0; h < 2; ++h) {
      size_t offA = ((size_t)bl * TT + t0 + srow + h * 64) * CC + kk + sseg;
      size_t offB = (size_t)(n0 + srow + h * 64) * CC + kk + sseg;
      gah[h] = *(const uint4*)(xhi + offA);
      gal[h] = *(const uint4*)(xlo + offA);
      gbh[h] = *(const uint4*)(whi + offB);
      gbl[h] = *(const uint4*)(wlo + offB);
    }
  };
  auto gstore = [&](int bufi) {
#pragma unroll
    for (int h = 0; h < 2; ++h) {
      int li = (srow + h * 64) * LDSTH + sseg;
      *(uint4*)&AsH[bufi][li] = gah[h];
      *(uint4*)&AsL[bufi][li] = gal[h];
      *(uint4*)&BsH[bufi][li] = gbh[h];
      *(uint4*)&BsL[bufi][li] = gbl[h];
    }
  };

  gload(0);
  gstore(0);
  __syncthreads();
  for (int i = 0; i < NK; ++i) {
    if (i + 1 < NK) gload((i + 1) * 32);  // latency hides behind compute
    compute(i & 1);
    if (i + 1 < NK) gstore((i + 1) & 1);  // disjoint buffer from readers
    __syncthreads();
  }
#endif

  // ---- epilogue: C/D VGPR layout -> z[b,t,n] fp32 ----
#pragma unroll
  for (int mt = 0; mt < 4; ++mt)
#pragma unroll
    for (int nt = 0; nt < 2; ++nt)
#pragma unroll
      for (int r = 0; r < 8; ++r) {
        int row = t0 + waveM * 64 + mt * 16 + r + lhalf * 8;
        int col = n0 + waveN * 32 + nt * 16 + lrow;
        z[((size_t)bg * TT + row) * NN + col] = acc[mt][nt][r];
      }
}

// ---------------------------------------------------------------------------
// Kernel 2: sequential CUBA-KWTA scan. Block = batch, thread = neuron.
// feedback = (1+a)*s_n - sum_j(s_j) + RBIAS;  out[b,n,t] = s_{t-1}.
// ---------------------------------------------------------------------------
__global__ __launch_bounds__(1024) void kwta_scan(
    const float* __restrict__ z, const float* __restrict__ se,
    float* __restrict__ out) {
  __shared__ float part[32];
  __shared__ float totalsh;

  const int n = threadIdx.x;
  const int b = blockIdx.x;
  const float a = fminf(fmaxf(se[0], 0.0f), 1.0f);
  const float onepa = 1.0f + a;
  const float RBIAS = (float)(NN - 200) / (float)NN;  // 0.8046875

  const float* zrow = z + (size_t)b * TT * NN + n;
  float* orow = out + ((size_t)b * NN + n) * TT;

  float u = 0.f, v = 0.f, s_prev = 0.f, sum_prev = 0.f;
  float zt = zrow[0];

  for (int t = 0; t < TT; ++t) {
    float znext = (t + 1 < TT) ? zrow[(size_t)(t + 1) * NN] : 0.f;
    float inp = zt + onepa * s_prev - sum_prev + RBIAS;
    u = 0.75f * u + inp;                   // (1-CD)*u + inp
    v = 0.90f * v + u;                     // (1-VD)*v + u
    float s = (v >= 1.0f) ? 1.0f : 0.0f;   // hard Heaviside forward
    v *= (1.0f - s);                       // reset where spiked
    orow[t] = s_prev;                      // one-step delay shift

    float wsum = s;
    wsum += __shfl_xor(wsum, 16, 32);
    wsum += __shfl_xor(wsum, 8, 32);
    wsum += __shfl_xor(wsum, 4, 32);
    wsum += __shfl_xor(wsum, 2, 32);
    wsum += __shfl_xor(wsum, 1, 32);
    if ((n & 31) == 0) part[n >> 5] = wsum;
    __syncthreads();
    if (n < 32) {
      float p = part[n];
      p += __shfl_xor(p, 16, 32);
      p += __shfl_xor(p, 8, 32);
      p += __shfl_xor(p, 4, 32);
      p += __shfl_xor(p, 2, 32);
      p += __shfl_xor(p, 1, 32);
      if (n == 0) totalsh = p;
    }
    __syncthreads();
    sum_prev = totalsh;
    s_prev = s;
    zt = znext;
  }
}

// ---------------------------------------------------------------------------
extern "C" void kernel_launch(void* const* d_in, const int* in_sizes, int n_in,
                              void* d_out, int out_size, void* d_ws, size_t ws_size,
                              hipStream_t stream) {
  const float* x  = (const float*)d_in[0];  // [64, 2048, 512]
  const float* w  = (const float*)d_in[1];  // [1024, 2048]
  const float* se = (const float*)d_in[2];  // [1]
  float* out = (float*)d_out;               // [64, 1024, 512]

  // Workspace layout: z fp32 | w_hi | w_lo | xT_hi(chunk) | xT_lo(chunk)
  const size_t zbytes  = (size_t)BB * TT * NN * 4;        // 128 MiB
  const size_t wplaneE = (size_t)NN * CC;
  const size_t xbatchE = (size_t)TT * CC;
  char* ws = (char*)d_ws;
  float* z = (float*)ws;
  unsigned short* whi = (unsigned short*)(ws + zbytes);
  unsigned short* wlo = whi + wplaneE;
  unsigned short* xhi = wlo + wplaneE;

  size_t fixed = zbytes + 2 * wplaneE * 2;
  size_t avail = (ws_size > fixed) ? (ws_size - fixed) : 0;
  int cb = (int)(avail / (2 * xbatchE * 2));
  if (cb < 1) cb = 1;
  if (cb > BB) cb = BB;
  unsigned short* xlo = xhi + (size_t)cb * xbatchE;

  hipLaunchKernelGGL(split_w, dim3((NN * CC) / (256 * 4)), dim3(256), 0, stream,
                     w, whi, wlo);

  for (int b0 = 0; b0 < BB; b0 += cb) {
    int nb = (b0 + cb <= BB) ? cb : (BB - b0);
    hipLaunchKernelGGL(xsplit, dim3(TT / 64, CC / 64, nb), dim3(256), 0, stream,
                       x, xhi, xlo, b0);
    hipLaunchKernelGGL(kwta_gemm_bf16x3, dim3(NN / 128, TT / 128, nb), dim3(256),
                       0, stream, xhi, xlo, whi, wlo, z, b0);
  }

  hipLaunchKernelGGL(kwta_scan, dim3(BB), dim3(1024), 0, stream, z, se, out);
}
